// ConformerEncoder_20306605375652
// MI455X (gfx1250) — matrix operand order repaired and checked by reading, not compiled, not run on hardware
//
#include <hip/hip_runtime.h>

// ---------------------------------------------------------------------------
// CDNA5 (gfx1250) Conformer encoder.
// GEMMs: v_wmma_f32_16x16x32_bf16, B staged via global_load_async_to_lds_b128
// (double buffered, s_wait_asynccnt), A software-pipelined in registers.
// Branch-free steady-state K loop (nstages is always even for this model).
// ---------------------------------------------------------------------------

typedef unsigned short u16;
typedef __attribute__((ext_vector_type(16))) __bf16 v16bf;
typedef __attribute__((ext_vector_type(8)))  float  v8f;

union FragBF { uint4 u4[2]; v16bf v; };

__device__ __forceinline__ u16 f2bf(float x) {
  unsigned int u = __float_as_uint(x);
  unsigned int r = (u + 0x7FFFu + ((u >> 16) & 1u)) >> 16;
  return (u16)r;
}
__device__ __forceinline__ float bf2f(u16 x) {
  return __uint_as_float(((unsigned int)x) << 16);
}
__device__ __forceinline__ float sigmoidf_(float x) {
  return 1.f / (1.f + __expf(-x));
}

// CDNA5 async global->LDS copy (16B per lane), tracked by ASYNCcnt.
__device__ __forceinline__ void async_ld16(unsigned lds_byte_addr,
                                           const u16* sbase, int goff_bytes) {
  asm volatile("global_load_async_to_lds_b128 %0, %1, %2"
               :: "v"(lds_byte_addr), "v"(goff_bytes), "s"(sbase)
               : "memory");
}
__device__ __forceinline__ void wait_async2() {
  asm volatile("s_wait_asynccnt 2" ::: "memory");
}
__device__ __forceinline__ void wait_async0() {
  asm volatile("s_wait_asynccnt 0" ::: "memory");
}

// ---------------- block reductions (wave32 + LDS) ----------------
__device__ __forceinline__ float blockReduceSum(float v, float* lds) {
  for (int m = 16; m; m >>= 1) v += __shfl_xor(v, m, 32);
  int lane = threadIdx.x & 31, w = threadIdx.x >> 5;
  __syncthreads();
  if (lane == 0) lds[w] = v;
  __syncthreads();
  int nw = blockDim.x >> 5;
  float s = 0.f;
  if ((int)threadIdx.x < nw) s = lds[threadIdx.x];
  if (w == 0) {
    for (int m = 16; m; m >>= 1) s += __shfl_xor(s, m, 32);
    if (lane == 0) lds[0] = s;
  }
  __syncthreads();
  float r = lds[0];
  __syncthreads();
  return r;
}
__device__ __forceinline__ float blockReduceMax(float v, float* lds) {
  for (int m = 16; m; m >>= 1) v = fmaxf(v, __shfl_xor(v, m, 32));
  int lane = threadIdx.x & 31, w = threadIdx.x >> 5;
  __syncthreads();
  if (lane == 0) lds[w] = v;
  __syncthreads();
  int nw = blockDim.x >> 5;
  float s = -3.4e38f;
  if ((int)threadIdx.x < nw) s = lds[threadIdx.x];
  if (w == 0) {
    for (int m = 16; m; m >>= 1) s = fmaxf(s, __shfl_xor(s, m, 32));
    if (lane == 0) lds[0] = s;
  }
  __syncthreads();
  float r = lds[0];
  __syncthreads();
  return r;
}

// ---------------- WMMA GEMM:  C[z] = act(A[z] * B[z]^T + bias) ----------------
// A: [M,K] bf16 row-major, lda elems, z-stride zA
// B: [N,K] bf16 row-major, ldb elems, z-stride zB      (i.e. C = A @ B^T)
// C: [M,N] (float or bf16), ldc elems, z-stride zC
// epilogue: c = acc + bias[n]; if SILU c = c*sigmoid(c); if res: c = res+alpha*c
// Requires: M % 128 == 0, N % 64 == 0, K % 64 == 0 (even #32-stages);
// 16B-aligned buffers.
//
// Block: 128 threads (4 waves), block tile 128Mx64N; wave tile 32Mx64N
// (8 accumulators). B tile (64x32 bf16, rows padded to 40 u16 to kill LDS
// bank conflicts) double-buffered in LDS via async DMA; pipeline is unrolled
// two stages per iteration with ping-pong A fragments (no register rotation).
#define BROW 40  // padded B row length in u16 (32 data + 8 pad)

__device__ __forceinline__ void loadA(const u16* ar, int kk, int halfsel,
                                      FragBF& f) {
  const u16* ap = ar + kk + halfsel * 8;
  f.u4[0] = *(const uint4*)(ap);
  f.u4[1] = *(const uint4*)(ap + 16);
}

template <bool OUT_BF16, bool SILU>
__global__ __launch_bounds__(128) void gemm_wmma(
    const u16* __restrict__ A, int lda, long long zA,
    const u16* __restrict__ B, int ldb, long long zB,
    void* __restrict__ Cv, int ldc, long long zC,
    const float* __restrict__ bias, const float* __restrict__ residual,
    float alpha, int M, int N, int K)
{
  __shared__ __align__(16) u16 Bs0[64 * BROW];
  __shared__ __align__(16) u16 Bs1[64 * BROW];

  const int z = blockIdx.z;
  A += (long long)z * zA;
  B += (long long)z * zB;
  float* Cf = (float*)Cv + (long long)z * zC;
  u16*   Cb = (u16*)Cv + (long long)z * zC;
  const float* R = residual ? (residual + (long long)z * zC) : nullptr;

  const int lane    = threadIdx.x & 31;
  const int wave    = threadIdx.x >> 5;
  const int halfsel = lane >> 4;   // 0: lanes 0-15, 1: lanes 16-31
  const int lrow    = lane & 15;

  const int m0 = blockIdx.y * 128 + wave * 32;
  const int n0 = blockIdx.x * 64;

  // Issue one B stage (64 rows x 32 K u16) into LDS at `dst` via async DMA.
  // 256 chunks of 8 u16 (16B); each of the 128 threads issues exactly 2.
  auto stageB = [&](int kk, u16* dst) {
    const int c0 = threadIdx.x * 2;
#pragma unroll
    for (int c = c0; c < c0 + 2; ++c) {
      const int n  = c >> 2;
      const int ko = (c & 3) * 8;
      const int goff = (int)(((long long)(n0 + n) * ldb + kk + ko) * 2);
      const unsigned laddr = (unsigned)(size_t)(dst + n * BROW + ko);
      async_ld16(laddr, B, goff);
    }
  };

  v8f acc[8] = {};

  // One compute stage: batch all 8 B-fragment LDS loads, then 8 WMMAs.
  auto computeStage = [&](const u16* bs, const FragBF& fa0, const FragBF& fa1) {
    FragBF b[4];
#pragma unroll
    for (int t = 0; t < 4; ++t) {
      const u16* bp = bs + (t * 16 + lrow) * BROW + halfsel * 8;
      b[t].u4[0] = *(const uint4*)(bp);
      b[t].u4[1] = *(const uint4*)(bp + 16);
    }
#pragma unroll
    for (int t = 0; t < 4; ++t) {
      acc[t] = __builtin_amdgcn_wmma_f32_16x16x32_bf16(
          false, fa0.v, false, b[t].v, (short)0, acc[t], false, false);
      acc[4 + t] = __builtin_amdgcn_wmma_f32_16x16x32_bf16(
          false, fa1.v, false, b[t].v, (short)0, acc[4 + t], false, false);
    }
  };

  const int nstages = K >> 5;   // always even, >= 2 for this model
  const u16* ar0 = A + (long long)(m0 + lrow) * lda;
  const u16* ar1 = A + (long long)(m0 + 16 + lrow) * lda;

  FragBF a00, a01, a10, a11;    // aXp: X = M-subtile, p = stage parity

  // Prologue: stage 0 -> Bs0, A(0) -> a*0
  stageB(0, Bs0);
  loadA(ar0, 0, halfsel, a00);
  loadA(ar1, 0, halfsel, a10);

  // Steady state: two stages per iteration, branch-free.
  for (int s = 0; s < nstages - 2; s += 2) {
    const int kk = s << 5;
    stageB(kk + 32, Bs1);
    loadA(ar0, kk + 32, halfsel, a01);
    loadA(ar1, kk + 32, halfsel, a11);
    wait_async2();               // stage s complete (2 newest outstanding)
    __syncthreads();
    computeStage(Bs0, a00, a10);
    __syncthreads();             // Bs0 free for re-staging

    stageB(kk + 64, Bs0);
    loadA(ar0, kk + 64, halfsel, a00);
    loadA(ar1, kk + 64, halfsel, a10);
    wait_async2();               // stage s+1 complete
    __syncthreads();
    computeStage(Bs1, a01, a11);
    __syncthreads();             // Bs1 free for re-staging
  }

  // Tail: stages nstages-2 (already staged in Bs0) and nstages-1.
  {
    const int kl = (nstages - 1) << 5;
    stageB(kl, Bs1);
    loadA(ar0, kl, halfsel, a01);
    loadA(ar1, kl, halfsel, a11);
    wait_async2();
    __syncthreads();
    computeStage(Bs0, a00, a10);
    __syncthreads();
    wait_async0();
    __syncthreads();
    computeStage(Bs1, a01, a11);
  }

#pragma unroll
  for (int i = 0; i < 2; ++i) {
#pragma unroll
    for (int t = 0; t < 4; ++t) {
      const int col = n0 + t * 16 + lrow;
      const float bv = bias ? bias[col] : 0.f;
#pragma unroll
      for (int r = 0; r < 8; ++r) {
        const int row = m0 + i * 16 + halfsel * 8 + r;
        float c = acc[i * 4 + t][r] + bv;
        if (SILU) c = c * sigmoidf_(c);
        const long long idx = (long long)row * ldc + col;
        if (R) c = R[idx] + alpha * c;
        if (OUT_BF16) Cb[idx] = f2bf(c);
        else          Cf[idx] = c;
      }
    }
  }
}

// ---------------- LayerNorm over last dim (=512), optional f32/bf16 outputs ----
__global__ __launch_bounds__(256) void ln_kernel(
    const float* __restrict__ in, const float* __restrict__ g,
    const float* __restrict__ b, float* __restrict__ outf, u16* __restrict__ outb)
{
  const int C = 512;
  const long long row = blockIdx.x;
  const float* x = in + row * C;
  const int t = threadIdx.x;
  float v0 = x[t], v1 = x[t + 256];
  __shared__ float lds[32];
  float s  = blockReduceSum(v0 + v1, lds);
  float ss = blockReduceSum(v0 * v0 + v1 * v1, lds);
  float mu  = s * (1.f / C);
  float var = ss * (1.f / C) - mu * mu;
  float rs  = rsqrtf(var + 1e-5f);
  float y0 = (v0 - mu) * rs * g[t] + b[t];
  float y1 = (v1 - mu) * rs * g[t + 256] + b[t + 256];
  if (outf) { outf[row * C + t] = y0; outf[row * C + t + 256] = y1; }
  if (outb) { outb[row * C + t] = f2bf(y0); outb[row * C + t + 256] = f2bf(y1); }
}

// ---------------- f32 -> bf16 (with zero-padding up to n_dst) ----------------
__global__ void f32_to_bf16(const float* __restrict__ src, u16* __restrict__ dst,
                            long long n_src, long long n_dst)
{
  long long i = (long long)blockIdx.x * 256 + threadIdx.x;
  if (i < n_dst) dst[i] = (i < n_src) ? f2bf(src[i]) : (u16)0;
}

// ---------------- QKV extraction: +pos_u/+pos_v, and V transposed -------------
__global__ __launch_bounds__(256) void extract_kernel(
    const float* __restrict__ qkv, const float* __restrict__ pu,
    const float* __restrict__ pv, u16* __restrict__ qu, u16* __restrict__ qv,
    u16* __restrict__ kk, u16* __restrict__ vT)
{
  const int S = 1024, H = 8, HD = 64;
  const long long m = blockIdx.x;          // 0..4095 = (b,s)
  const int bb = (int)(m >> 10), s = (int)(m & 1023);
  const float* row = qkv + m * 1536;
  for (int c = threadIdx.x; c < 512; c += blockDim.x) {
    const int h = c >> 6, dd = c & 63;
    const float q = row[h * 192 + dd];
    const float k = row[h * 192 + 64 + dd];
    const float v = row[h * 192 + 128 + dd];
    const long long base = (((long long)bb * H + h) * S + s) * HD + dd;
    qu[base] = f2bf(q + pu[c]);
    qv[base] = f2bf(q + pv[c]);
    kk[base] = f2bf(k);
    vT[(((long long)bb * H + h) * HD + dd) * S + s] = f2bf(v);
  }
}

// ---------------- attention softmax with fused rel_shift ---------------------
// shifted_bd[q,k] = bd[q, k + 1023 - q]  (always in range for q,k in [0,1024))
__global__ __launch_bounds__(256) void softmax_kernel(
    const u16* __restrict__ ac, const u16* __restrict__ bd,
    u16* __restrict__ out, float scale)
{
  const int S = 1024, P = 2048;
  const int q = blockIdx.x, h = blockIdx.y;
  const u16* a = ac + ((long long)h * S + q) * S;
  const u16* d = bd + ((long long)h * S + q) * P + (1023 - q);
  u16* o = out + ((long long)h * S + q) * S;
  __shared__ float lds[32];
  float vals[4];
  float mx = -3.4e38f;
#pragma unroll
  for (int i = 0; i < 4; ++i) {
    const int k = threadIdx.x + i * 256;
    const float v = (bf2f(a[k]) + bf2f(d[k])) * scale;
    vals[i] = v;
    mx = fmaxf(mx, v);
  }
  mx = blockReduceMax(mx, lds);
  float se = 0.f;
#pragma unroll
  for (int i = 0; i < 4; ++i) { vals[i] = __expf(vals[i] - mx); se += vals[i]; }
  se = blockReduceSum(se, lds);
  const float inv = 1.f / se;
#pragma unroll
  for (int i = 0; i < 4; ++i) o[threadIdx.x + i * 256] = f2bf(vals[i] * inv);
}

// ---------------- GLU over channel split (cols 0..511 vs 512..1023) ----------
__global__ void glu_kernel(const float* __restrict__ z, float* __restrict__ o,
                           long long n)
{
  long long i = (long long)blockIdx.x * 256 + threadIdx.x;
  if (i >= n) return;
  const long long m = i >> 9;
  const int c = (int)(i & 511);
  const float a = z[m * 1024 + c];
  const float g = z[m * 1024 + 512 + c];
  o[i] = a * sigmoidf_(g);
}

// ---------------- depthwise conv, kernel 31, zero pad, per batch of 1024 -----
__global__ __launch_bounds__(256) void dwconv_kernel(
    const float* __restrict__ x, const float* __restrict__ w,
    const float* __restrict__ bias, float* __restrict__ o)
{
  const int S = 1024, C = 512, KK = 31;
  const long long m = blockIdx.x;
  const int bt = (int)(m >> 10), t = (int)(m & 1023);
  for (int c = threadIdx.x; c < C; c += blockDim.x) {
    float acc = bias[c];
#pragma unroll
    for (int j = 0; j < KK; ++j) {
      const int tt = t - 15 + j;
      if (tt >= 0 && tt < S)
        acc += w[c * KK + j] * x[(((long long)bt << 10) + tt) * C + c];
    }
    o[m * C + c] = acc;
  }
}

// ---------------- BatchNorm stats over (b,t) per channel ---------------------
__global__ __launch_bounds__(256) void bnstats_kernel(
    const float* __restrict__ y, float* __restrict__ mean, float* __restrict__ var)
{
  const int C = 512;
  const long long R = 4096;
  const int c = blockIdx.x;
  float s = 0.f, ss = 0.f;
  for (long long r = threadIdx.x; r < R; r += blockDim.x) {
    const float v = y[r * C + c];
    s += v; ss += v * v;
  }
  __shared__ float lds[32];
  s  = blockReduceSum(s, lds);
  ss = blockReduceSum(ss, lds);
  if (threadIdx.x == 0) {
    const float mu = s / (float)R;
    mean[c] = mu;
    var[c]  = ss / (float)R - mu * mu;
  }
}

__global__ void bnsilu_kernel(const float* __restrict__ y,
                              const float* __restrict__ mean,
                              const float* __restrict__ var,
                              const float* __restrict__ g,
                              const float* __restrict__ b,
                              u16* __restrict__ o, long long n)
{
  long long i = (long long)blockIdx.x * 256 + threadIdx.x;
  if (i >= n) return;
  const int c = (int)(i & 511);
  const float v = (y[i] - mean[c]) * rsqrtf(var[c] + 1e-5f) * g[c] + b[c];
  o[i] = f2bf(v * sigmoidf_(v));
}

// ===========================================================================
extern "C" void kernel_launch(void* const* d_in, const int* in_sizes, int n_in,
                              void* d_out, int out_size, void* d_ws, size_t ws_size,
                              hipStream_t stream)
{
  (void)in_sizes; (void)n_in; (void)out_size; (void)ws_size;
  const int Bn = 4, S = 1024, D = 512, F = 2048, H = 8, HD = 64;
  const int M = Bn * S, P2 = 2048;
  const long long PR = 2 * S - 1;   // 2047
  const float scale = 0.04419417382415922f;  // 1/sqrt(512)

  const float* src     = (const float*)d_in[0];
  const float* pos     = (const float*)d_in[1];
  const float* final_b = (const float*)d_in[2];
  const float* final_g = (const float*)d_in[3];
  // per-layer params, jax pytree (alphabetical) order:
  //  0 attn_in_w  1 attn_out_b  2 attn_out_w  3 bn_b  4 bn_g  5 bott_b
  //  6 bott_w  7 conv_ln_b  8 conv_ln_g  9 dw_b 10 dw_w 11 ffn1_b1 12 ffn1_b2
  // 13 ffn1_ln_b 14 ffn1_ln_g 15 ffn1_w1 16 ffn1_w2 17 ffn2_b1 18 ffn2_b2
  // 19 ffn2_ln_b 20 ffn2_ln_g 21 ffn2_w1 22 ffn2_w2 23 norm1_b 24 norm1_g
  // 25 norm2_b 26 norm2_g 27 pos_u 28 pos_v 29 pos_w 30 pw_b 31 pw_w
  auto LP = [&](int l, int i) { return (const float*)d_in[4 + l * 32 + i]; };

  char* base = (char*)d_ws;
  size_t off = 0;
  auto alloc = [&](size_t bytes) -> void* {
    void* p = base + off;
    off = (off + bytes + 255) & ~(size_t)255;
    return p;
  };

  float* xA   = (float*)alloc((size_t)M * D * 4);
  float* xB   = (float*)alloc((size_t)M * D * 4);
  u16* lnb    = (u16*)alloc((size_t)M * D * 2);
  u16* mid    = (u16*)alloc((size_t)M * F * 2);
  float* qkv  = (float*)alloc((size_t)M * 3 * D * 4);
  u16* posb   = (u16*)alloc((size_t)P2 * D * 2);
  u16* pkb    = (u16*)alloc((size_t)P2 * D * 2);
  u16* qu     = (u16*)alloc((size_t)M * D * 2);
  u16* qv     = (u16*)alloc((size_t)M * D * 2);
  u16* kb     = (u16*)alloc((size_t)M * D * 2);
  u16* vT     = (u16*)alloc((size_t)M * D * 2);
  u16* acb    = (u16*)alloc((size_t)H * S * S * 2);
  u16* bdb    = (u16*)alloc((size_t)H * S * P2 * 2);
  u16* atb    = (u16*)alloc((size_t)H * S * S * 2);
  u16* ctxb   = (u16*)alloc((size_t)M * D * 2);
  float* cnvz = (float*)alloc((size_t)M * 2 * D * 4);
  float* glub = (float*)alloc((size_t)M * D * 4);
  float* dwo  = (float*)alloc((size_t)M * D * 4);
  float* bnm  = (float*)alloc((size_t)D * 4);
  float* bnv  = (float*)alloc((size_t)D * 4);
  u16* bnb    = (u16*)alloc((size_t)M * D * 2);
  u16* w_in   = (u16*)alloc((size_t)3 * D * D * 2);
  u16* w_out  = (u16*)alloc((size_t)D * D * 2);
  u16* w_pos  = (u16*)alloc((size_t)D * D * 2);
  u16* w_bot  = (u16*)alloc((size_t)2 * D * D * 2);
  u16* w_pw   = (u16*)alloc((size_t)D * D * 2);
  u16* w_f11  = (u16*)alloc((size_t)F * D * 2);
  u16* w_f12  = (u16*)alloc((size_t)F * D * 2);
  u16* w_f21  = (u16*)alloc((size_t)F * D * 2);
  u16* w_f22  = (u16*)alloc((size_t)F * D * 2);

  auto cvt = [&](const float* s, u16* dst, long long ns, long long nd) {
    f32_to_bf16<<<dim3((unsigned)((nd + 255) / 256)), dim3(256), 0, stream>>>(s, dst, ns, nd);
  };
  auto ln = [&](const float* in, const float* g, const float* b, float* of, u16* ob) {
    ln_kernel<<<dim3(M), dim3(256), 0, stream>>>(in, g, b, of, ob);
  };
  auto gemm = [&](const u16* A, int lda, long long zA, const u16* Bp, int ldb,
                  long long zB, void* C, int ldc, long long zC, const float* bias,
                  const float* res, float alpha, int Mi, int Ni, int Ki, int Z,
                  int obf, int silu) {
    dim3 g(Ni / 64, Mi / 128, Z), blk(128);
    if (obf) {
      if (silu)
        gemm_wmma<true, true><<<g, blk, 0, stream>>>(A, lda, zA, Bp, ldb, zB, C,
                                                     ldc, zC, bias, res, alpha, Mi, Ni, Ki);
      else
        gemm_wmma<true, false><<<g, blk, 0, stream>>>(A, lda, zA, Bp, ldb, zB, C,
                                                      ldc, zC, bias, res, alpha, Mi, Ni, Ki);
    } else {
      gemm_wmma<false, false><<<g, blk, 0, stream>>>(A, lda, zA, Bp, ldb, zB, C,
                                                     ldc, zC, bias, res, alpha, Mi, Ni, Ki);
    }
  };

  // pos embeddings -> bf16, zero-padded row 2047 -> 2048
  cvt(pos, posb, PR * D, (long long)P2 * D);

  const float* cur = src;
  auto other = [&]() -> float* { return (cur == xA) ? xB : xA; };

  for (int l = 0; l < 2; ++l) {
    // weights -> bf16
    cvt(LP(l, 0),  w_in,  (long long)3 * D * D, (long long)3 * D * D);
    cvt(LP(l, 2),  w_out, (long long)D * D,     (long long)D * D);
    cvt(LP(l, 29), w_pos, (long long)D * D,     (long long)D * D);
    cvt(LP(l, 6),  w_bot, (long long)2 * D * D, (long long)2 * D * D);
    cvt(LP(l, 31), w_pw,  (long long)D * D,     (long long)D * D);
    cvt(LP(l, 15), w_f11, (long long)F * D,     (long long)F * D);
    cvt(LP(l, 16), w_f12, (long long)F * D,     (long long)F * D);
    cvt(LP(l, 21), w_f21, (long long)F * D,     (long long)F * D);
    cvt(LP(l, 22), w_f22, (long long)F * D,     (long long)F * D);

    // ---- macaron FFN 1: x += 0.5 * ffn(ln(x)) ----
    ln(cur, LP(l, 14), LP(l, 13), nullptr, lnb);
    gemm(lnb, D, 0, w_f11, D, 0, mid, F, 0, LP(l, 11), nullptr, 0.f,
         M, F, D, 1, 1, 1);                               // bf16 out, SiLU
    { float* t = other();
      gemm(mid, F, 0, w_f12, F, 0, t, D, 0, LP(l, 12), cur, 0.5f,
           M, D, F, 1, 0, 0);
      cur = t; }

    // ---- rel-pos MHA: x = skip + mha(ln(x)) ----
    const float* skip = cur;
    ln(cur, LP(l, 24), LP(l, 23), nullptr, lnb);
    gemm(lnb, D, 0, w_in, D, 0, qkv, 3 * D, 0, nullptr, nullptr, 0.f,
         M, 3 * D, D, 1, 0, 0);
    gemm(posb, D, 0, w_pos, D, 0, pkb, D, 0, nullptr, nullptr, 0.f,
         P2, D, D, 1, 1, 0);                              // pk in bf16
    extract_kernel<<<dim3(M), dim3(256), 0, stream>>>(qkv, LP(l, 27), LP(l, 28),
                                                      qu, qv, kb, vT);
    for (int b = 0; b < Bn; ++b) {
      const u16* quB = qu + (long long)b * H * S * HD;
      const u16* qvB = qv + (long long)b * H * S * HD;
      const u16* kbB = kb + (long long)b * H * S * HD;
      const u16* vTB = vT + (long long)b * H * HD * S;
      gemm(quB, HD, (long long)S * HD, kbB, HD, (long long)S * HD,
           acb, S, (long long)S * S, nullptr, nullptr, 0.f,
           S, S, HD, H, 1, 0);                            // ac = q_u @ k^T
      gemm(qvB, HD, (long long)S * HD, pkb, D, 64,
           bdb, P2, (long long)S * P2, nullptr, nullptr, 0.f,
           S, P2, HD, H, 1, 0);                           // bd = q_v @ pk^T
      softmax_kernel<<<dim3(S, H), dim3(256), 0, stream>>>(acb, bdb, atb, scale);
      gemm(atb, S, (long long)S * S, vTB, S, (long long)HD * S,
           ctxb + (long long)b * S * D, D, 64, nullptr, nullptr, 0.f,
           S, HD, S, H, 1, 0);                            // ctx = attn @ V
    }
    { float* t = other();
      gemm(ctxb, D, 0, w_out, D, 0, t, D, 0, LP(l, 1), skip, 1.0f,
           M, D, D, 1, 0, 0);
      cur = t; }

    // ---- conv module: x += conv(ln(x)) ----
    ln(cur, LP(l, 8), LP(l, 7), nullptr, lnb);
    gemm(lnb, D, 0, w_bot, D, 0, cnvz, 2 * D, 0, LP(l, 5), nullptr, 0.f,
         M, 2 * D, D, 1, 0, 0);
    glu_kernel<<<dim3((unsigned)(((long long)M * D + 255) / 256)), dim3(256), 0,
                 stream>>>(cnvz, glub, (long long)M * D);
    dwconv_kernel<<<dim3(M), dim3(256), 0, stream>>>(glub, LP(l, 10), LP(l, 9), dwo);
    bnstats_kernel<<<dim3(D), dim3(256), 0, stream>>>(dwo, bnm, bnv);
    bnsilu_kernel<<<dim3((unsigned)(((long long)M * D + 255) / 256)), dim3(256), 0,
                    stream>>>(dwo, bnm, bnv, LP(l, 4), LP(l, 3), bnb, (long long)M * D);
    { float* t = other();
      gemm(bnb, D, 0, w_pw, D, 0, t, D, 0, LP(l, 30), cur, 1.0f,
           M, D, D, 1, 0, 0);
      cur = t; }

    // ---- macaron FFN 2 ----
    ln(cur, LP(l, 20), LP(l, 19), nullptr, lnb);
    gemm(lnb, D, 0, w_f21, D, 0, mid, F, 0, LP(l, 17), nullptr, 0.f,
         M, F, D, 1, 1, 1);
    { float* t = other();
      gemm(mid, F, 0, w_f22, F, 0, t, D, 0, LP(l, 18), cur, 0.5f,
           M, D, F, 1, 0, 0);
      cur = t; }

    // ---- norm2 ----
    { float* t = other();
      ln(cur, LP(l, 26), LP(l, 25), t, nullptr);
      cur = t; }
  }

  // final layer norm -> output (fp32)
  ln_kernel<<<dim3(M), dim3(256), 0, stream>>>(cur, final_g, final_b,
                                               (float*)d_out, (u16*)nullptr);
}
